// SparseGatedMoE_66331474919464
// MI455X (gfx1250) — matrix operand order repaired and checked by reading, not compiled, not run on hardware
//
#include <hip/hip_runtime.h>
#include <stdint.h>

#define B_TOK 8192
#define D_DIM 512
#define E_NUM 8
#define H_DIM 2048
#define M_TILE 64
#define HB 128

// strides (in u16 elements): every b128 access 16B-aligned, lanes spread over banks
#define XS_S 520   // 64 x (512 + 8)
#define W1_S 40    // 128 x (32 + 8)   (x2 buffers)
#define W2_S 40    // 512 x (32 + 8)   (x2 buffers)
#define HT_S 136   // 64 x (128 + 8)

typedef __attribute__((ext_vector_type(16))) __bf16 v16bf;
typedef __attribute__((ext_vector_type(8)))  float  v8f;
typedef __attribute__((ext_vector_type(4)))  unsigned int u32x4;
typedef __attribute__((ext_vector_type(8)))  int i32x8;
typedef __attribute__((ext_vector_type(4)))  int i32x4;

#if defined(__has_builtin)
#if __has_builtin(__builtin_amdgcn_tensor_load_to_lds)
#define HAVE_TDM 1
#endif
#endif
#ifndef HAVE_TDM
#define HAVE_TDM 0
#endif

union FragU { uint4 u4[2]; v16bf v; };

__device__ __forceinline__ unsigned short f2bf(float f) {
  unsigned int u = __float_as_uint(f);
  u += 0x7FFFu + ((u >> 16) & 1u);   // round-to-nearest-even
  return (unsigned short)(u >> 16);
}

__device__ __forceinline__ unsigned lds_off(const void* p) {
  return (unsigned)(uintptr_t)p;     // low 32 bits of generic ptr = LDS byte offset
}

// per-lane 16B async DMA: global -> LDS, tracked by ASYNCcnt, no VGPR data
__device__ __forceinline__ void async_copy_b128(unsigned lds, const void* gaddr) {
  asm volatile("global_load_async_to_lds_b128 %0, %1, off"
               :: "v"(lds), "v"(gaddr) : "memory");
}
__device__ __forceinline__ void wait_async0() {
  asm volatile("s_wait_asynccnt 0x0" ::: "memory");
}

#if HAVE_TDM
// TDM: DMA a 2D strided tile (data_size=2B) into LDS with 16B padding every 64B,
// matching the 40-u16 padded LDS row stride. Issued by one wave, TENSORcnt-tracked.
__device__ __forceinline__ void tdm_load_2d(unsigned lds, const void* gptr,
                                            int tile_d0, int tile_d1,
                                            int tens_d0, int tens_d1, int stride_d0) {
  unsigned long long ga = (unsigned long long)(uintptr_t)gptr;
  u32x4 g0;
  g0[0] = 1u;                                             // count=1 (valid), user mode
  g0[1] = lds;                                            // lds_addr
  g0[2] = (unsigned)(ga & 0xFFFFFFFFu);                   // global_addr[31:0]
  g0[3] = (unsigned)((ga >> 32) & 0x1FFFFFFu) | (2u << 30); // addr[56:32] | type=2
  i32x8 g1;
  g1[0] = (1 << 16)        // data_size = 2 bytes
        | (1 << 20)        // pad_enable
        | (3 << 22)        // pad_interval: every 16 DWORDs (64B)
        | (3 << 25);       // pad_amount: 4 DWORDs (16B)
  g1[1] = (tens_d0 & 0xFFFF) << 16;                         // tensor_dim0[15:0]
  g1[2] = ((tens_d0 >> 16) & 0xFFFF) | ((tens_d1 & 0xFFFF) << 16);
  g1[3] = ((tens_d1 >> 16) & 0xFFFF) | ((tile_d0 & 0xFFFF) << 16);
  g1[4] = (tile_d1 & 0xFFFF);                               // tile_dim1, tile_dim2=0
  g1[5] = stride_d0;                                        // tensor_dim0_stride
  g1[6] = 0;
  g1[7] = 0;
  i32x4 z4 = {0, 0, 0, 0};
  i32x8 z8 = {0, 0, 0, 0, 0, 0, 0, 0};
  // 6-arg form (clang-23 / therock-10.0 headers): groups 2,3 unused for 2D tiles
  __builtin_amdgcn_tensor_load_to_lds(g0, g1, z4, z4, z8, 0);
}
#endif

// A fragment (16x32 bf16, row-major LDS): lanes0-15 row m0+L K k0+[0..7],[16..23];
// lanes16-31 same rows K k0+[8..15],[24..31]
__device__ __forceinline__ v16bf frag_a(const unsigned short* base, int m0, int k0,
                                        int stride, int lane) {
  FragU f;
  int row = m0 + (lane & 15);
  int kb  = k0 + ((lane >> 4) << 3);
  f.u4[0] = *(const uint4*)(base + row * stride + kb);
  f.u4[1] = *(const uint4*)(base + row * stride + kb + 16);
  return f.v;
}

// B fragment (32x16 bf16) from [col][k]-contiguous LDS tile:
// lanes0-15 col n0+L K [0..7],[16..23]; lanes16-31 col n0+L-16 K [8..15],[24..31]
__device__ __forceinline__ v16bf frag_b(const unsigned short* base, int n0,
                                        int stride, int lane) {
  FragU f;
  int col = n0 + (lane & 15);
  int kb  = (lane >> 4) << 3;
  f.u4[0] = *(const uint4*)(base + col * stride + kb);
  f.u4[1] = *(const uint4*)(base + col * stride + kb + 16);
  return f.v;
}

// ---------------------------------------------------------------- utilities
__global__ void zero_kernel(float* __restrict__ y, int* __restrict__ counts) {
  int i = blockIdx.x * blockDim.x + threadIdx.x;
  float4 z = make_float4(0.f, 0.f, 0.f, 0.f);
  ((float4*)y)[i] = z;
  if (blockIdx.x == 0 && threadIdx.x < E_NUM) counts[threadIdx.x] = 0;
}

__global__ void cvt_x(const float* __restrict__ x, unsigned short* __restrict__ xbf) {
  int i = blockIdx.x * blockDim.x + threadIdx.x;
  float2 v = ((const float2*)x)[i];
  unsigned int p = (unsigned int)f2bf(v.x) | ((unsigned int)f2bf(v.y) << 16);
  ((unsigned int*)xbf)[i] = p;
}

// in: [E][R][C] fp32 -> out: [E][C][R] bf16
__global__ void transpose_cvt(const float* __restrict__ in, unsigned short* __restrict__ out,
                              int R, int C) {
  __shared__ float tile[32][33];
  int e  = blockIdx.z;
  int c0 = blockIdx.x * 32, r0 = blockIdx.y * 32;
  int tx = threadIdx.x & 31, ty = threadIdx.x >> 5;
  const float* src = in + (size_t)e * R * C;
  unsigned short* dst = out + (size_t)e * R * C;
  #pragma unroll
  for (int j = 0; j < 32; j += 8)
    tile[ty + j][tx] = src[(size_t)(r0 + ty + j) * C + c0 + tx];
  __syncthreads();
  #pragma unroll
  for (int j = 0; j < 32; j += 8)
    dst[(size_t)(c0 + ty + j) * R + r0 + tx] = f2bf(tile[tx][ty + j]);
}

// ---------------------------------------------------------------- gating
__global__ void gate_kernel(const float* __restrict__ x, const float* __restrict__ wg,
                            int* __restrict__ counts, int* __restrict__ list,
                            float* __restrict__ gates) {
  int b = blockIdx.x * blockDim.x + threadIdx.x;
  const float* xr = x + (size_t)b * D_DIM;
  float acc[E_NUM];
  #pragma unroll
  for (int e = 0; e < E_NUM; ++e) acc[e] = 0.f;
  for (int d = 0; d < D_DIM; ++d) {
    float xv = xr[d];
    #pragma unroll
    for (int e = 0; e < E_NUM; ++e) acc[e] = fmaf(xv, wg[d * E_NUM + e], acc[e]);
  }
  int i0 = 0; float v0 = acc[0];
  #pragma unroll
  for (int e = 1; e < E_NUM; ++e) if (acc[e] > v0) { v0 = acc[e]; i0 = e; }
  int i1 = -1; float v1 = -3.402823466e38f;
  #pragma unroll
  for (int e = 0; e < E_NUM; ++e) if (e != i0 && acc[e] > v1) { v1 = acc[e]; i1 = e; }
  float e1 = expf(v1 - v0);
  float inv = 1.f / (1.f + e1);
  float g0 = inv, g1 = e1 * inv;
  int s0 = atomicAdd(&counts[i0], 1);
  list[i0 * B_TOK + s0] = b;  gates[i0 * B_TOK + s0] = g0;
  int s1 = atomicAdd(&counts[i1], 1);
  list[i1 * B_TOK + s1] = b;  gates[i1 * B_TOK + s1] = g1;
}

// ---------------------------------------------------------------- fused expert MLP
struct __align__(16) Smem {
  unsigned short Xs[M_TILE * XS_S];        // 66.5 KB
  unsigned short W1s[2 * HB * W1_S];       // 20 KB  (double-buffered)
  unsigned short W2s[2 * D_DIM * W2_S];    // 80 KB  (double-buffered)
  unsigned short Ht[M_TILE * HT_S];        // 17 KB
  float b1c[HB];
  float b2c[D_DIM];
  int   tok[M_TILE];
  float gvals[M_TILE];
};

__device__ __forceinline__ void issue_w1(Smem& sm, const unsigned short* w1e,
                                         int buf, int hb, int k0, int tid) {
  #pragma unroll
  for (int j = 0; j < 2; ++j) {           // 512 b128 moves, 2 per thread
    int i = j * 256 + tid;
    int row = i >> 2, q = i & 3;
    const unsigned short* g = w1e + (size_t)(hb * HB + row) * D_DIM + k0 + q * 8;
    async_copy_b128(lds_off(sm.W1s + buf * (HB * W1_S) + row * W1_S + q * 8), g);
  }
}

__device__ __forceinline__ void issue_w2(Smem& sm, const unsigned short* w2e,
                                         int hoff, int buf, int tid, int wave) {
#if HAVE_TDM
  if (wave == 0)
    tdm_load_2d(lds_off(sm.W2s + buf * (D_DIM * W2_S)), w2e + hoff,
                /*tile*/32, D_DIM, /*tensor*/H_DIM, D_DIM, /*stride*/H_DIM);
#else
  #pragma unroll
  for (int j = 0; j < 8; ++j) {           // 2048 b128 moves, 8 per thread
    int i = j * 256 + tid;
    int row = i >> 2, q = i & 3;
    const unsigned short* g = w2e + (size_t)row * H_DIM + hoff + q * 8;
    async_copy_b128(lds_off(sm.W2s + buf * (D_DIM * W2_S) + row * W2_S + q * 8), g);
  }
#endif
}

__device__ __forceinline__ void wait_w2() {
#if HAVE_TDM
  __builtin_amdgcn_s_wait_tensorcnt(0);
#else
  wait_async0();
#endif
}

__global__ void __launch_bounds__(256, 1) moe_kernel(
    const unsigned short* __restrict__ xbf,   // [B][D] bf16
    const unsigned short* __restrict__ w1t,   // [E][H][D] bf16 (W1^T)
    const unsigned short* __restrict__ w2t,   // [E][D][H] bf16 (W2^T)
    const float* __restrict__ b1,
    const float* __restrict__ b2,
    const int* __restrict__ counts,
    const int* __restrict__ list,
    const float* __restrict__ gates,
    float* __restrict__ y) {
  __shared__ Smem sm;
  const int e    = blockIdx.y;
  const int cnt  = counts[e];
  const int row0 = blockIdx.x * M_TILE;
  if (row0 >= cnt) return;                     // uniform early exit, before barriers

  const int tid  = threadIdx.x;
  const int lane = tid & 31;
  const int wave = tid >> 5;

  if (tid < M_TILE) {
    int s = row0 + tid;
    sm.tok[tid]   = (s < cnt) ? list[e * B_TOK + s]  : 0;
    sm.gvals[tid] = (s < cnt) ? gates[e * B_TOK + s] : 0.f;
  }
  sm.b2c[tid]       = b2[e * D_DIM + tid];
  sm.b2c[tid + 256] = b2[e * D_DIM + tid + 256];
  __syncthreads();

  const unsigned short* w1e = w1t + (size_t)e * H_DIM * D_DIM;
  const unsigned short* w2e = w2t + (size_t)e * D_DIM * H_DIM;

  // async gather of the 64 routed token rows into Xs (16 b128 per thread)
  #pragma unroll
  for (int j = 0; j < 16; ++j) {
    int i = j * 256 + tid;
    int row = i >> 5, seg = i & 31;
    const unsigned short* g = xbf + (size_t)sm.tok[row] * D_DIM + seg * 8;
    async_copy_b128(lds_off(sm.Xs + row * XS_S + seg * 8), g);
  }
  issue_w1(sm, w1e, 0, 0, 0, tid);             // preload first W1 chunk

  v8f zero8 = {};
  v8f oacc[4][4];
  #pragma unroll
  for (int mt = 0; mt < 4; ++mt)
    #pragma unroll
    for (int nt = 0; nt < 4; ++nt) oacc[mt][nt] = zero8;

  for (int hb = 0; hb < H_DIM / HB; ++hb) {
    if (tid < HB) sm.b1c[tid] = b1[e * H_DIM + hb * HB + tid];
#if HAVE_TDM
    issue_w2(sm, w2e, hb * HB, 0, tid, wave);  // TDM prefetch overlaps all of stage 1
#endif

    // ---- stage 1: Htile[64,128] = X[64,512] @ W1[:, chunk]
    v8f hacc[4];
    #pragma unroll
    for (int mt = 0; mt < 4; ++mt) hacc[mt] = zero8;

    for (int i1 = 0; i1 < D_DIM / 32; ++i1) {
      const int k0 = i1 * 32;
      wait_async0();                           // chunk i1 landed (and Xs on first pass)
      __syncthreads();
      if (i1 + 1 < D_DIM / 32)       issue_w1(sm, w1e, (i1 + 1) & 1, hb, k0 + 32, tid);
      else if (hb + 1 < H_DIM / HB)  issue_w1(sm, w1e, 0, hb + 1, 0, tid);
      const unsigned short* W1b = sm.W1s + (i1 & 1) * (HB * W1_S);
      v16bf bf = frag_b(W1b, wave * 16, W1_S, lane);
      #pragma unroll
      for (int mt = 0; mt < 4; ++mt) {
        v16bf af = frag_a(sm.Xs, mt * 16, k0, XS_S, lane);
        hacc[mt] = __builtin_amdgcn_wmma_f32_16x16x32_bf16(
            false, af, false, bf, (short)0, hacc[mt], false, false);
      }
    }

    // bias + exact GELU, write bf16 Htile
    #pragma unroll
    for (int mt = 0; mt < 4; ++mt) {
      #pragma unroll
      for (int r = 0; r < 8; ++r) {
        int ml = mt * 16 + r + ((lane >> 4) << 3);
        int nl = wave * 16 + (lane & 15);
        float v = hacc[mt][r] + sm.b1c[nl];
        v = 0.5f * v * (1.f + erff(v * 0.70710678118654752f));
        sm.Ht[ml * HT_S + nl] = f2bf(v);
      }
    }
    __syncthreads();
#if !HAVE_TDM
    issue_w2(sm, w2e, hb * HB, 0, tid, wave);
#endif

    // ---- stage 2: out[64,512] += Htile @ W2[chunk, :]
    for (int i2 = 0; i2 < HB / 32; ++i2) {
      wait_w2();
      __syncthreads();
      if (i2 + 1 < HB / 32)
        issue_w2(sm, w2e, hb * HB + (i2 + 1) * 32, (i2 + 1) & 1, tid, wave);
      const unsigned short* W2b = sm.W2s + (i2 & 1) * (D_DIM * W2_S);
      v16bf afh[4];
      #pragma unroll
      for (int mt = 0; mt < 4; ++mt) afh[mt] = frag_a(sm.Ht, mt * 16, i2 * 32, HT_S, lane);
      #pragma unroll
      for (int nt = 0; nt < 4; ++nt) {
        v16bf bf2 = frag_b(W2b, wave * 64 + nt * 16, W2_S, lane);
        #pragma unroll
        for (int mt = 0; mt < 4; ++mt)
          oacc[mt][nt] = __builtin_amdgcn_wmma_f32_16x16x32_bf16(
              false, afh[mt], false, bf2, (short)0, oacc[mt][nt], false, false);
      }
    }
  }
  __syncthreads();

  // epilogue: +b2, gate scale, deterministic 2-way atomic combine
  #pragma unroll
  for (int mt = 0; mt < 4; ++mt) {
    #pragma unroll
    for (int r = 0; r < 8; ++r) {
      int m = mt * 16 + r + ((lane >> 4) << 3);
      float g = sm.gvals[m];
      if (g == 0.f) continue;
      int t = sm.tok[m];
      #pragma unroll
      for (int nt = 0; nt < 4; ++nt) {
        int d = wave * 64 + nt * 16 + (lane & 15);
        float v = (oacc[mt][nt][r] + sm.b2c[d]) * g;
        atomicAdd(&y[(size_t)t * D_DIM + d], v);
      }
    }
  }
}

// ---------------------------------------------------------------- launch
extern "C" void kernel_launch(void* const* d_in, const int* in_sizes, int n_in,
                              void* d_out, int out_size, void* d_ws, size_t ws_size,
                              hipStream_t stream) {
  const float* x  = (const float*)d_in[0];
  const float* wg = (const float*)d_in[1];
  const float* W1 = (const float*)d_in[2];
  const float* b1 = (const float*)d_in[3];
  const float* W2 = (const float*)d_in[4];
  const float* b2 = (const float*)d_in[5];
  float* y = (float*)d_out;

  uintptr_t p = (uintptr_t)d_ws;
  auto alloc = [&](size_t bytes) {
    uintptr_t r = (p + 255) & ~(uintptr_t)255; p = r + bytes; return (void*)r;
  };
  int*   counts = (int*)  alloc(E_NUM * sizeof(int));
  int*   list   = (int*)  alloc((size_t)E_NUM * B_TOK * sizeof(int));
  float* gates  = (float*)alloc((size_t)E_NUM * B_TOK * sizeof(float));
  unsigned short* xbf = (unsigned short*)alloc((size_t)B_TOK * D_DIM * 2);
  unsigned short* w1t = (unsigned short*)alloc((size_t)E_NUM * H_DIM * D_DIM * 2);
  unsigned short* w2t = (unsigned short*)alloc((size_t)E_NUM * D_DIM * H_DIM * 2);
  (void)ws_size; (void)in_sizes; (void)n_in; (void)out_size;

  zero_kernel<<<(B_TOK * D_DIM / 4) / 256, 256, 0, stream>>>(y, counts);
  cvt_x<<<(B_TOK * D_DIM / 2) / 256, 256, 0, stream>>>(x, xbf);
  transpose_cvt<<<dim3(H_DIM / 32, D_DIM / 32, E_NUM), 256, 0, stream>>>(W1, w1t, D_DIM, H_DIM);
  transpose_cvt<<<dim3(D_DIM / 32, H_DIM / 32, E_NUM), 256, 0, stream>>>(W2, w2t, H_DIM, D_DIM);
  gate_kernel<<<B_TOK / 256, 256, 0, stream>>>(x, wg, counts, list, gates);
  moe_kernel<<<dim3(B_TOK / M_TILE, E_NUM), 256, 0, stream>>>(
      xbf, w1t, w2t, b1, b2, counts, list, gates, y);
}
